// GATLinkPredictor_88149908783544
// MI455X (gfx1250) — matrix operand order repaired and checked by reading, not compiled, not run on hardware
//
#include <hip/hip_runtime.h>
#include <hip/hip_bf16.h>

#define NEG_SLOPE 0.2f

typedef __attribute__((ext_vector_type(16))) __bf16 v16bf;
typedef __attribute__((ext_vector_type(8)))  float  v8f;
typedef unsigned short ubf;  // raw bf16 storage

union fragu { v16bf v; uint4 q[2]; };

static __device__ __forceinline__ ubf f2bf_raw(float f) {
    union { float f; unsigned u; } x; x.f = f;
    unsigned r = (x.u + 0x7FFFu + ((x.u >> 16) & 1u)) >> 16;  // RNE
    return (ubf)r;
}

// Order-preserving float <-> uint encoding for atomicMax-based segment max.
static __device__ __forceinline__ unsigned fenc(float f) {
    unsigned u = __float_as_uint(f);
    return (u & 0x80000000u) ? ~u : (u | 0x80000000u);
}
static __device__ __forceinline__ float fdec(unsigned u) {
    unsigned v = (u & 0x80000000u) ? (u & 0x7FFFFFFFu) : ~u;
    return __uint_as_float(v);
}

// ---------------------------------------------------------------------------
// Precision converters (run once per GEMM input; tiny vs edge traffic)
// ---------------------------------------------------------------------------
__global__ void f32_to_bf16_k(const float* __restrict__ src, ubf* __restrict__ dst, long n4) {
    long t = (long)blockIdx.x * blockDim.x + threadIdx.x;
    if (t >= n4) return;
    float4 f = *(const float4*)(src + t * 4);
    union { ubf u[4]; uint2 d; } o;
    o.u[0] = f2bf_raw(f.x); o.u[1] = f2bf_raw(f.y);
    o.u[2] = f2bf_raw(f.z); o.u[3] = f2bf_raw(f.w);
    *(uint2*)(dst + t * 4) = o.d;
}

// B[K,N] f32 -> Bt[N,K] bf16 (weights are tiny: <=128x128)
__global__ void w_transpose_bf16_k(const float* __restrict__ B, ubf* __restrict__ Bt,
                                   int K, int N) {
    int t = blockIdx.x * blockDim.x + threadIdx.x;
    if (t >= K * N) return;
    int n = t / K, k = t % K;
    Bt[t] = f2bf_raw(B[(size_t)k * N + n]);
}

// ---------------------------------------------------------------------------
// WMMA GEMM: C[M,N] = A[M,K] @ B[K,N]; A bf16 row-major, B as Bt[N,K] bf16.
// One wave32 per 16x64 strip: one A fragment feeds 4 WMMAs per 32-K step.
// Fragments are pure b128 vector loads (layouts per ISA 7.12.2):
//   A lane frag: K = {aoff..aoff+7, 16+aoff..16+aoff+7}, aoff = (lane&16)?8:0
//   B lane frag: K = {boff..boff+7, 16+boff..16+boff+7} of column bn, boff = (lane&16)?8:0
// M%16==0, N%64==0, K%32==0 (holds for all calls here).
// ---------------------------------------------------------------------------
__global__ void wmma_gemm_bf16(const ubf* __restrict__ A,
                               const ubf* __restrict__ Bt,
                               float* __restrict__ C,
                               int M, int N, int K) {
    int wave = (int)((blockIdx.x * blockDim.x + threadIdx.x) >> 5);
    int lane = threadIdx.x & 31;
    int stripsN = N >> 6;
    int total = (M >> 4) * stripsN;
    if (wave >= total) return;            // uniform per wave; EXEC stays all-1s
    int tm = wave / stripsN, tn = wave % stripsN;

    int am   = (tm << 4) + (lane & 15);
    int aoff = (lane & 16) ? 8 : 0;
    int boff = (lane & 16) ? 8 : 0;
    const ubf* Arow = A + (size_t)am * K;

    v8f acc[4] = {};
    for (int k0 = 0; k0 < K; k0 += 32) {
        fragu a;
        a.q[0] = *(const uint4*)(Arow + k0 + aoff);
        a.q[1] = *(const uint4*)(Arow + k0 + 16 + aoff);
#pragma unroll
        for (int u = 0; u < 4; ++u) {
            int bn = (tn << 6) + (u << 4) + (lane & 15);
            const ubf* Brow = Bt + (size_t)bn * K + k0 + boff;
            fragu b;
            b.q[0] = *(const uint4*)(Brow);
            b.q[1] = *(const uint4*)(Brow + 16);
            acc[u] = __builtin_amdgcn_wmma_f32_16x16x32_bf16(
                false, a.v, false, b.v, (short)0, acc[u], false, false);
        }
    }

    int rbase = (tm << 4) + ((lane & 16) ? 8 : 0);
#pragma unroll
    for (int u = 0; u < 4; ++u) {
        int cn = (tn << 6) + (u << 4) + (lane & 15);
#pragma unroll
        for (int r = 0; r < 8; ++r)
            C[(size_t)(rbase + r) * N + cn] = acc[u][r];
    }
}

// ---------------------------------------------------------------------------
// Small helper kernels
// ---------------------------------------------------------------------------
__global__ void compute_kedge(const float* __restrict__ We1, const float* __restrict__ ae1,
                              const float* __restrict__ We2, const float* __restrict__ ae2,
                              float* __restrict__ kedge) {
    if (blockIdx.x == 0 && threadIdx.x == 0) {
        for (int h = 0; h < 2; ++h) {
            float s = 0.f;
            for (int c = 0; c < 64; ++c) s += We1[h * 64 + c] * ae1[h * 64 + c];
            kedge[h] = s;
        }
        float s = 0.f;
        for (int c = 0; c < 64; ++c) s += We2[c] * ae2[c];
        kedge[2] = s;
    }
}

__global__ void deg_wsum_k(const int* __restrict__ col, const float* __restrict__ w,
                           float* __restrict__ deg, float* __restrict__ wsum, int E) {
    int t = blockIdx.x * blockDim.x + threadIdx.x;
    if (t >= E) return;
    int c = col[t];
    atomicAdd(&deg[c], 1.0f);
    atomicAdd(&wsum[c], w[t]);
}

__global__ void loopw_k(const float* __restrict__ deg, const float* __restrict__ wsum,
                        float* __restrict__ lw, int Nn) {
    int t = blockIdx.x * blockDim.x + threadIdx.x;
    if (t >= Nn) return;
    float d = deg[t];
    lw[t] = (d > 0.f) ? (wsum[t] / d) : 0.f;
}

// Per-node attention projections: asrc[n,h] = sum_c h[n,h,c]*att_src[h,c], same for dst.
__global__ void node_attn_k(const float* __restrict__ h,
                            const float* __restrict__ att_src,
                            const float* __restrict__ att_dst,
                            float* __restrict__ asrc, float* __restrict__ adst,
                            int Nn, int H, int C) {
    int t = blockIdx.x * blockDim.x + threadIdx.x;
    if (t >= Nn * H) return;
    int n = t / H, hh = t % H;
    const float* hp = h + (size_t)n * H * C + (size_t)hh * C;
    float s1 = 0.f, s2 = 0.f;
    for (int c = 0; c < C; ++c) {
        float v = hp[c];
        s1 += v * att_src[hh * C + c];
        s2 += v * att_dst[hh * C + c];
    }
    asrc[t] = s1;
    adst[t] = s2;
}

// Pass 1: alpha = leaky_relu(asrc[row]+adst[col]+w*k_h); segment max via atomicMax(enc).
__global__ void edge_alpha_max_k(const int* __restrict__ row, const int* __restrict__ col,
                                 const float* __restrict__ w, const float* __restrict__ lw,
                                 const float* __restrict__ asrc, const float* __restrict__ adst,
                                 const float* __restrict__ kedge, int kofs,
                                 float* __restrict__ alpha, unsigned* __restrict__ menc,
                                 int E, int Nn, int H) {
    long t = (long)blockIdx.x * blockDim.x + threadIdx.x;
    long total = (long)(E + Nn) * H;
    if (t >= total) return;
    int e = (int)(t / H), hh = (int)(t % H);
    int r, c; float ww;
    if (e < E) { r = row[e]; c = col[e]; ww = w[e]; }
    else       { r = c = e - E; ww = lw[r]; }
    float a = asrc[r * H + hh] + adst[c * H + hh] + ww * kedge[kofs + hh];
    a = (a > 0.f) ? a : NEG_SLOPE * a;
    alpha[t] = a;
    atomicMax(&menc[(size_t)c * H + hh], fenc(a));
}

// Pass 2: ea = exp(alpha - m[col]); segment sum via atomicAdd.
__global__ void edge_exp_sum_k(const int* __restrict__ col, const unsigned* __restrict__ menc,
                               float* __restrict__ alpha, float* __restrict__ s,
                               int E, int Nn, int H) {
    long t = (long)blockIdx.x * blockDim.x + threadIdx.x;
    long total = (long)(E + Nn) * H;
    if (t >= total) return;
    int e = (int)(t / H), hh = (int)(t % H);
    int c = (e < E) ? col[e] : (e - E);
    float ea = __expf(alpha[t] - fdec(menc[(size_t)c * H + hh]));
    alpha[t] = ea;
    atomicAdd(&s[(size_t)c * H + hh], ea);
}

// Pass 3: out[col] += h[row] * (ea / (s[col]+eps)); one thread per (edge, 4 channels).
__global__ void edge_aggregate_k(const int* __restrict__ row, const int* __restrict__ col,
                                 const float* __restrict__ ea, const float* __restrict__ s,
                                 const float* __restrict__ hsrc, float* __restrict__ out,
                                 int E, int Nn, int H, int C) {
    int G = (H * C) >> 2;
    long t = (long)blockIdx.x * blockDim.x + threadIdx.x;
    long total = (long)(E + Nn) * G;
    if (t >= total) return;
    int e = (int)(t / G), g = (int)(t % G);
    int r, c;
    if (e < E) { r = row[e]; c = col[e]; }
    else       { r = c = e - E; }
    int ch = g << 2;
    int hh = ch / C;
    float coef = ea[(size_t)e * H + hh] / (s[(size_t)c * H + hh] + 1e-16f);
    const float4 hv = *(const float4*)(hsrc + (size_t)r * H * C + ch);
    float* op = out + (size_t)c * H * C + ch;
    atomicAdd(op + 0, hv.x * coef);
    atomicAdd(op + 1, hv.y * coef);
    atomicAdd(op + 2, hv.z * coef);
    atomicAdd(op + 3, hv.w * coef);
}

__global__ void bias_act_k(float* __restrict__ out, const float* __restrict__ b,
                           int Nn, int HC, int relu) {
    long t = (long)blockIdx.x * blockDim.x + threadIdx.x;
    if (t >= (long)Nn * HC) return;
    float v = out[t] + b[(int)(t % HC)];
    out[t] = relu ? fmaxf(v, 0.f) : v;
}

__global__ void decode_dot_k(const float* __restrict__ z, const int* __restrict__ eli,
                             float* __restrict__ out, int L, int C) {
    int t = blockIdx.x * blockDim.x + threadIdx.x;
    if (t >= L) return;
    int i = eli[t], j = eli[L + t];
    const float4* zi = (const float4*)(z + (size_t)i * C);
    const float4* zj = (const float4*)(z + (size_t)j * C);
    float s = 0.f;
#pragma unroll
    for (int q = 0; q < 16; ++q) {  // C == 64
        float4 a = zi[q], b = zj[q];
        s += a.x * b.x + a.y * b.y + a.z * b.z + a.w * b.w;
    }
    out[t] = s;
}

// ---------------------------------------------------------------------------
// Host launcher
// ---------------------------------------------------------------------------
static inline int cdiv(long a, long b) { return (int)((a + b - 1) / b); }

extern "C" void kernel_launch(void* const* d_in, const int* in_sizes, int n_in,
                              void* d_out, int out_size, void* d_ws, size_t ws_size,
                              hipStream_t stream) {
    const int IN_CH = 128, HID = 64, HEADS = 2, OUT_CH = 64;
    const int N  = in_sizes[0] / IN_CH;   // 50000
    const int E  = in_sizes[2];           // 1.6M
    const int L  = in_sizes[3] / 2;       // 100000
    const int ET = E + N;

    const float* x   = (const float*)d_in[0];
    const int*   ei  = (const int*)d_in[1];
    const int*   row = ei;
    const int*   col = ei + E;
    const float* ew  = (const float*)d_in[2];
    const int*   eli = (const int*)d_in[3];
    const float* W1  = (const float*)d_in[4];
    const float* as1 = (const float*)d_in[5];
    const float* ad1 = (const float*)d_in[6];
    const float* We1 = (const float*)d_in[7];
    const float* ae1 = (const float*)d_in[8];
    const float* b1  = (const float*)d_in[9];
    const float* W2  = (const float*)d_in[10];
    const float* as2 = (const float*)d_in[11];
    const float* ad2 = (const float*)d_in[12];
    const float* We2 = (const float*)d_in[13];
    const float* ae2 = (const float*)d_in[14];
    const float* b2  = (const float*)d_in[15];

    // Workspace layout (f32 units, 4-float aligned)
    float* ws = (float*)d_ws;
    size_t o = 0;
    auto alloc = [&](size_t n) { float* p = ws + o; o += (n + 3) & ~(size_t)3; return p; };
    float* deg    = alloc(N);
    float* wsum   = alloc(N);
    float* lw     = alloc(N);
    float* kedge  = alloc(4);
    float* h1     = alloc((size_t)N * 128);
    float* asrc1  = alloc((size_t)N * 2);
    float* adst1  = alloc((size_t)N * 2);
    float* alpha1 = alloc((size_t)ET * 2);
    float* m1     = alloc((size_t)N * 2);
    float* s1     = alloc((size_t)N * 2);
    float* out1   = alloc((size_t)N * 128);
    float* h2     = alloc((size_t)N * 64);
    float* asrc2  = alloc(N);
    float* adst2  = alloc(N);
    float* alpha2 = alloc(ET);
    float* m2     = alloc(N);
    float* s2     = alloc(N);
    float* out2   = alloc((size_t)N * 64);
    // bf16 staging (element count / 2 in f32 units)
    ubf* xbf   = (ubf*)alloc((size_t)N * 64);   // [N,128] bf16
    ubf* h1rbf = (ubf*)alloc((size_t)N * 64);   // [N,128] bf16
    ubf* W1t   = (ubf*)alloc(128 * 128 / 2);    // [128,128] bf16 (W1^T)
    ubf* W2t   = (ubf*)alloc(64 * 128 / 2);     // [64,128]  bf16 (W2^T)
    (void)ws_size;

    // Zero accumulators (menc=0 encodes below every real float; every node has a self-loop)
    hipMemsetAsync(deg,  0, (size_t)N * 2 * sizeof(float), stream);  // deg + wsum
    hipMemsetAsync(m1,   0, (size_t)N * 2 * sizeof(float), stream);
    hipMemsetAsync(s1,   0, (size_t)N * 2 * sizeof(float), stream);
    hipMemsetAsync(out1, 0, (size_t)N * 128 * sizeof(float), stream);
    hipMemsetAsync(m2,   0, (size_t)N * sizeof(float), stream);
    hipMemsetAsync(s2,   0, (size_t)N * sizeof(float), stream);
    hipMemsetAsync(out2, 0, (size_t)N * 64 * sizeof(float), stream);

    const int BT = 256;

    // One-time precision staging for WMMA
    f32_to_bf16_k<<<cdiv((long)N * 128 / 4, BT), BT, 0, stream>>>(x, xbf, (long)N * 128 / 4);
    w_transpose_bf16_k<<<cdiv(128 * 128, BT), BT, 0, stream>>>(W1, W1t, 128, 128);
    w_transpose_bf16_k<<<cdiv(128 * 64, BT), BT, 0, stream>>>(W2, W2t, 128, 64);

    // Self-loop weights (fill_value='mean')
    compute_kedge<<<1, 32, 0, stream>>>(We1, ae1, We2, ae2, kedge);
    deg_wsum_k<<<cdiv(E, BT), BT, 0, stream>>>(col, ew, deg, wsum, E);
    loopw_k<<<cdiv(N, BT), BT, 0, stream>>>(deg, wsum, lw, N);

    // ----- Layer 1: heads=2, C=64, concat -----
    {
        long waves = (long)(N / 16) * (128 / 64);
        wmma_gemm_bf16<<<cdiv(waves * 32, 128), 128, 0, stream>>>(xbf, W1t, h1, N, 128, 128);
    }
    node_attn_k<<<cdiv((long)N * HEADS, BT), BT, 0, stream>>>(h1, as1, ad1, asrc1, adst1, N, HEADS, HID);
    edge_alpha_max_k<<<cdiv((long)ET * HEADS, BT), BT, 0, stream>>>(
        row, col, ew, lw, asrc1, adst1, kedge, 0, alpha1, (unsigned*)m1, E, N, HEADS);
    edge_exp_sum_k<<<cdiv((long)ET * HEADS, BT), BT, 0, stream>>>(
        col, (const unsigned*)m1, alpha1, s1, E, N, HEADS);
    edge_aggregate_k<<<cdiv((long)ET * 32, BT), BT, 0, stream>>>(
        row, col, alpha1, s1, h1, out1, E, N, HEADS, HID);
    bias_act_k<<<cdiv((long)N * 128, BT), BT, 0, stream>>>(out1, b1, N, 128, 1);

    // ----- Layer 2: heads=1, C=64, mean (==identity for 1 head) -----
    f32_to_bf16_k<<<cdiv((long)N * 128 / 4, BT), BT, 0, stream>>>(out1, h1rbf, (long)N * 128 / 4);
    {
        long waves = (long)(N / 16) * (64 / 64);
        wmma_gemm_bf16<<<cdiv(waves * 32, 128), 128, 0, stream>>>(h1rbf, W2t, h2, N, 64, 128);
    }
    node_attn_k<<<cdiv(N, BT), BT, 0, stream>>>(h2, as2, ad2, asrc2, adst2, N, 1, OUT_CH);
    edge_alpha_max_k<<<cdiv((long)ET, BT), BT, 0, stream>>>(
        row, col, ew, lw, asrc2, adst2, kedge, 2, alpha2, (unsigned*)m2, E, N, 1);
    edge_exp_sum_k<<<cdiv((long)ET, BT), BT, 0, stream>>>(
        col, (const unsigned*)m2, alpha2, s2, E, N, 1);
    edge_aggregate_k<<<cdiv((long)ET * 16, BT), BT, 0, stream>>>(
        row, col, alpha2, s2, h2, out2, E, N, 1, OUT_CH);
    bias_act_k<<<cdiv((long)N * 64, BT), BT, 0, stream>>>(out2, b2, N, 64, 0);

    // ----- Decode: dot(z[src], z[dst]) -----
    decode_dot_k<<<cdiv(L, BT), BT, 0, stream>>>(out2, eli, (float*)d_out, L, OUT_CH);
}
